// PointPillarsModel_4243427688687
// MI455X (gfx1250) — matrix operand order repaired and checked by reading, not compiled, not run on hardware
//
#include <hip/hip_runtime.h>
#include <hip/hip_bf16.h>

// ---------- types ----------
typedef __attribute__((ext_vector_type(16))) __bf16 v16bf;
typedef __attribute__((ext_vector_type(8)))  float  v8f;
typedef __attribute__((ext_vector_type(8)))  unsigned short u16x8;

union FragBF { u16x8 h[2]; v16bf v; };

__device__ __forceinline__ unsigned short f2bf(float f) {
  unsigned int u = __float_as_uint(f);
  u += 0x7fffu + ((u >> 16) & 1u);            // round-to-nearest-even
  return (unsigned short)(u >> 16);
}
__device__ __forceinline__ float bf2f(unsigned short h) {
  return __uint_as_float(((unsigned int)h) << 16);
}

// async global->LDS copy of 16 bytes (per lane), tracked by ASYNCcnt
__device__ __forceinline__ void async_copy_b128(unsigned int lds_byte_off,
                                                const void* gptr) {
  asm volatile("global_load_async_to_lds_b128 %0, %1, off"
               :: "v"(lds_byte_off), "v"((unsigned long long)(uintptr_t)gptr)
               : "memory");
}
__device__ __forceinline__ void wait_async0() {
  asm volatile("s_wait_asynccnt 0x0" ::: "memory");
}
__device__ __forceinline__ unsigned int lds_off32(const void* p) {
  return (unsigned int)(uintptr_t)p;          // low 32 bits = LDS byte offset
}

// ---------- problem constants ----------
#define BN_   4
#define P_    12000
#define NPT_  32
#define H0_   666
#define W0_   666
#define H1_   333
#define W1_   333
#define H2_   166
#define W2_   166
#define SEG_  64
#define EPS_  1e-5f

// =====================================================================
// Weight conversion: (Cout,Cin,3,3) f32 ->
//   blocked bf16 [cichunk][cotile][tap][co(64)][ci(32)] (fully contiguous
//   per (cichunk,cotile) so the conv kernel can async-copy it to LDS)
// =====================================================================
__global__ void convert_w(const float* __restrict__ src, unsigned short* __restrict__ dst,
                          int Cout, int Cin) {
  int idx = blockIdx.x * blockDim.x + threadIdx.x;
  int total = Cout * Cin * 9;
  if (idx >= total) return;
  int tap = idx % 9;
  int t   = idx / 9;
  int ci_g = t % Cin;
  int co_g = t / Cin;
  int cic = ci_g >> 5, ci = ci_g & 31;
  int nt  = co_g >> 6, co = co_g & 63;
  int ntiles = Cout >> 6;
  size_t d = ((((size_t)cic * ntiles + nt) * 9 + tap) * 64 + co) * 32 + ci;
  dst[d] = f2bf(src[idx]);
}

// heads: concat box(21)|cls(3)|dir(2) -> padded 32 ch,
// blocked layout [cichunk][tap][co(32)][ci(32)]
__global__ void convert_heads(const float* __restrict__ bw, const float* __restrict__ cw,
                              const float* __restrict__ dw, const float* __restrict__ bb,
                              const float* __restrict__ cb, const float* __restrict__ db,
                              unsigned short* __restrict__ wt, float* __restrict__ hb) {
  int idx = blockIdx.x * blockDim.x + threadIdx.x;   // over 9*32*256
  if (idx >= 9 * 32 * 256) return;
  int ci_g = idx % 256;
  int co   = (idx / 256) % 32;
  int tap  = idx / (256 * 32);
  float v = 0.f;
  if      (co < 21) v = bw[((size_t)(co      * 256 + ci_g)) * 9 + tap];
  else if (co < 24) v = cw[((size_t)((co-21) * 256 + ci_g)) * 9 + tap];
  else if (co < 26) v = dw[((size_t)((co-24) * 256 + ci_g)) * 9 + tap];
  size_t d = ((((size_t)(ci_g >> 5) * 9 + tap) * 32 + co) * 32) + (ci_g & 31);
  wt[d] = f2bf(v);
  if (idx < 32) {
    float b = 0.f;
    if      (idx < 21) b = bb[idx];
    else if (idx < 24) b = cb[idx - 21];
    else if (idx < 26) b = db[idx - 24];
    hb[idx] = b;
  }
}

// =====================================================================
// Stage 0: pointwise linear (K=9 -> 64) -> bf16 rows buffer
// =====================================================================
__global__ __launch_bounds__(256) void linear_kernel(const float* __restrict__ pillars,
                                                     const float* __restrict__ lw,
                                                     unsigned short* __restrict__ xbuf,
                                                     long rows) {
  __shared__ float w[64 * 9];
  for (int i = threadIdx.x; i < 576; i += 256) w[i] = lw[i];
  __syncthreads();
  long row = (long)blockIdx.x * 256 + threadIdx.x;
  if (row >= rows) return;
  float p[9];
  #pragma unroll
  for (int d = 0; d < 9; ++d) p[d] = pillars[row * 9 + d];
  #pragma unroll 4
  for (int co = 0; co < 64; ++co) {
    float s = 0.f;
    #pragma unroll
    for (int d = 0; d < 9; ++d) s += p[d] * w[co * 9 + d];
    xbuf[row * 64 + co] = f2bf(s);
  }
}

// =====================================================================
// Deterministic 2-stage channel statistics (mean / inv-std)
// =====================================================================
__global__ __launch_bounds__(256) void stats_partial_rows(const unsigned short* __restrict__ in,
                                                          float* __restrict__ part, long n) {
  const int c = blockIdx.x, seg = blockIdx.y, tid = threadIdx.x;
  const long per = (n + SEG_ - 1) / SEG_;
  const long lo = (long)seg * per;
  const long hi = (lo + per < n) ? (lo + per) : n;
  float s = 0.f, q = 0.f;
  for (long i = lo + tid; i < hi; i += 256) {
    float v = bf2f(in[i * 64 + c]);
    s += v; q += v * v;
  }
  __shared__ float r1[256], r2[256];
  r1[tid] = s; r2[tid] = q; __syncthreads();
  for (int off = 128; off > 0; off >>= 1) {
    if (tid < off) { r1[tid] += r1[tid + off]; r2[tid] += r2[tid + off]; }
    __syncthreads();
  }
  if (!tid) { part[(c * SEG_ + seg) * 2] = r1[0]; part[(c * SEG_ + seg) * 2 + 1] = r2[0]; }
}

__global__ __launch_bounds__(256) void stats_partial_nchw(const unsigned short* __restrict__ in,
                                                          float* __restrict__ part,
                                                          int C, int HW, int Bn) {
  const int c = blockIdx.x, seg = blockIdx.y, tid = threadIdx.x;
  const long n = (long)Bn * HW;
  const long per = (n + SEG_ - 1) / SEG_;
  const long lo = (long)seg * per;
  const long hi = (lo + per < n) ? (lo + per) : n;
  float s = 0.f, q = 0.f;
  for (long i = lo + tid; i < hi; i += 256) {
    int bb = (int)(i / HW), p = (int)(i % HW);
    float v = bf2f(in[((size_t)bb * C + c) * HW + p]);
    s += v; q += v * v;
  }
  __shared__ float r1[256], r2[256];
  r1[tid] = s; r2[tid] = q; __syncthreads();
  for (int off = 128; off > 0; off >>= 1) {
    if (tid < off) { r1[tid] += r1[tid + off]; r2[tid] += r2[tid + off]; }
    __syncthreads();
  }
  if (!tid) { part[(c * SEG_ + seg) * 2] = r1[0]; part[(c * SEG_ + seg) * 2 + 1] = r2[0]; }
}

__global__ __launch_bounds__(64) void stats_finalize(const float* __restrict__ part,
                                                     float* __restrict__ stats, float count) {
  const int c = blockIdx.x, tid = threadIdx.x;
  __shared__ float r1[64], r2[64];
  r1[tid] = part[(c * SEG_ + tid) * 2];
  r2[tid] = part[(c * SEG_ + tid) * 2 + 1];
  __syncthreads();
  for (int off = 32; off > 0; off >>= 1) {
    if (tid < off) { r1[tid] += r1[tid + off]; r2[tid] += r2[tid + off]; }
    __syncthreads();
  }
  if (!tid) {
    float m = r1[0] / count;
    float var = r2[0] / count - m * m;
    stats[2 * c] = m;
    stats[2 * c + 1] = rsqrtf(var + EPS_);
  }
}

// =====================================================================
// BN + ReLU + max over the 32 points of a pillar -> feats (bf16)
// =====================================================================
__global__ __launch_bounds__(256) void bnmax_kernel(const unsigned short* __restrict__ x,
                                                    const float* __restrict__ stats,
                                                    const float* __restrict__ g,
                                                    const float* __restrict__ bta,
                                                    unsigned short* __restrict__ feats,
                                                    int total) {
  int idx = blockIdx.x * 256 + threadIdx.x;
  if (idx >= total) return;
  int c = idx & 63;
  long bp = idx >> 6;
  float m = stats[2 * c], is = stats[2 * c + 1], gg = g[c], bb = bta[c];
  float best = 0.f;                                   // relu => max >= 0
  for (int n = 0; n < NPT_; ++n) {
    float v = bf2f(x[((bp * NPT_ + n) << 6) + c]);
    v = fmaxf((v - m) * is * gg + bb, 0.f);
    best = fmaxf(best, v);
  }
  feats[idx] = f2bf(best);
}

// =====================================================================
// zero + scatter into pseudo image (B,64,H0,W0) bf16
// =====================================================================
__global__ void zero_u32(unsigned int* __restrict__ p, size_t n) {
  size_t i = (size_t)blockIdx.x * blockDim.x + threadIdx.x;
  size_t stride = (size_t)gridDim.x * blockDim.x;
  for (; i < n; i += stride) p[i] = 0u;
}

__global__ __launch_bounds__(256) void scatter_kernel(const unsigned short* __restrict__ feats,
                                                      const int* __restrict__ coords,
                                                      unsigned short* __restrict__ pseudo,
                                                      int total) {
  int idx = blockIdx.x * 256 + threadIdx.x;
  if (idx >= total) return;
  int c = idx & 63;
  int bp = idx >> 6;
  int b = bp / P_;
  int y = coords[bp * 4 + 1];
  int x = coords[bp * 4 + 2];
  if ((unsigned)y < (unsigned)H0_ && (unsigned)x < (unsigned)W0_)
    pseudo[(((size_t)b * 64 + c) * H0_ + y) * W0_ + x] = feats[idx];
}

// =====================================================================
// 3x3 SAME conv, implicit GEMM on v_wmma_f32_16x16x32_bf16.
//   block tile: 32 output pixels (one row) x 64 output channels, 8 waves.
//   Per ci-chunk: stage 3x34x32 input halo once (reused by all 9 taps)
//   + async-copy the 9-tap weight block, then 9 unrolled WMMAs.
// =====================================================================
template<int CIN>
__global__ __launch_bounds__(256) void conv3x3_wmma(const unsigned short* __restrict__ in,
                                                    const unsigned short* __restrict__ wt,
                                                    unsigned short* __restrict__ out,
                                                    int Cout, int H, int W) {
  __shared__ __align__(16) unsigned short sIn[3 * 36 * 40];  // [row][pix<34][ci] +pad
  __shared__ __align__(16) unsigned short sB[9 * 64 * 40];   // [tap][co][ci]     +pad

  const int tid   = threadIdx.x;
  const int lane  = tid & 31;
  const int wave  = tid >> 5;
  const int waveM = wave & 1;
  const int waveN = wave >> 1;
  const int l16   = lane & 15;
  const int lhalf = lane >> 4;

  const int xbase  = blockIdx.x * 32;
  const int y      = blockIdx.y;
  const int ntiles = Cout >> 6;
  const int b      = blockIdx.z / ntiles;
  const int nt     = blockIdx.z % ntiles;
  const int cobase = nt << 6;

  v8f acc = {0.f, 0.f, 0.f, 0.f, 0.f, 0.f, 0.f, 0.f};

  #pragma unroll 1
  for (int cic = 0; cic < (CIN >> 5); ++cic) {
    const int c0 = cic << 5;

    // ---- stage A halo: 3 rows x 34 pixels x 32 ci (transposed to ci-major)
    for (int idx = tid; idx < 3 * 34 * 32; idx += 256) {
      const int row = idx / (34 * 32);
      const int rem = idx % (34 * 32);
      const int ci  = rem / 34;
      const int pix = rem % 34;          // consecutive tid -> consecutive pix
      const int iy  = y + row - 1;
      const int gx  = xbase + pix - 1;
      unsigned short v = 0;
      if ((unsigned)iy < (unsigned)H && (unsigned)gx < (unsigned)W)
        v = in[(((size_t)b * CIN + (c0 + ci)) * H + iy) * W + gx];
      sIn[(row * 36 + pix) * 40 + ci] = v;
    }

    // ---- stage B: contiguous 9*64*32 bf16 weight block via async b128 copies
    {
      const unsigned short* wsrc = wt + ((size_t)(cic * ntiles + nt)) * (9 * 64 * 32);
      #pragma unroll
      for (int it = 0; it < 9; ++it) {                // 2304 groups / 256 threads
        const int g   = it * 256 + tid;
        const int tap = g >> 8;                       // g / 256
        const int r   = g & 255;
        const int co  = r >> 2;
        const int cib = (r & 3) << 3;
        async_copy_b128(lds_off32(&sB[(tap * 64 + co) * 40 + cib]),
                        wsrc + ((size_t)(tap * 64 + co) * 32 + cib));
      }
      wait_async0();
    }
    __syncthreads();

    // ---- 9 taps, all from staged LDS ----
    #pragma unroll
    for (int tap = 0; tap < 9; ++tap) {
      const int dy = tap / 3;          // 0..2
      const int dx = tap % 3;          // 0..2
      FragBF a, bm;
      const int pixIdx = waveM * 16 + l16 + dx;            // 0..33
      const int abase  = (dy * 36 + pixIdx) * 40 + lhalf * 8;
      a.h[0] = *(const u16x8*)&sIn[abase];
      a.h[1] = *(const u16x8*)&sIn[abase + 16];
      const int cor = waveN * 16 + l16;
      const int kb  = lhalf * 8;
      bm.h[0] = *(const u16x8*)&sB[(tap * 64 + cor) * 40 + kb];
      bm.h[1] = *(const u16x8*)&sB[(tap * 64 + cor) * 40 + kb + 16];
      acc = __builtin_amdgcn_wmma_f32_16x16x32_bf16(false, a.v, false, bm.v,
                                                    (short)0, acc, false, false);
    }
    __syncthreads();
  }

  // D layout: VGPR r, lane L -> M = r + (L<16?0:8), N = L%16
  const int co = cobase + waveN * 16 + l16;
  #pragma unroll
  for (int r = 0; r < 8; ++r) {
    const int x = xbase + waveM * 16 + r + lhalf * 8;
    if (x < W)
      out[(((size_t)b * Cout + co) * H + y) * W + x] = f2bf(acc[r]);
  }
}

// =====================================================================
// fused BN + ReLU (+ optional 2x2 VALID maxpool)
// =====================================================================
__global__ __launch_bounds__(256) void bn_relu_pool(const unsigned short* __restrict__ in,
                                                    const float* __restrict__ stats,
                                                    const float* __restrict__ g,
                                                    const float* __restrict__ bta,
                                                    unsigned short* __restrict__ out,
                                                    int C, int Hin, int Win, int Hout, int Wout,
                                                    int total) {
  int idx = blockIdx.x * 256 + threadIdx.x;
  if (idx >= total) return;
  int wo = idx % Wout;
  int t = idx / Wout;
  int ho = t % Hout; t /= Hout;
  int c = t % C;
  int b = t / C;
  float m = stats[2 * c], is = stats[2 * c + 1], gg = g[c], bb = bta[c];
  size_t base = (((size_t)b * C + c) * Hin + 2 * ho) * Win + 2 * wo;
  float best = 0.f;
  #pragma unroll
  for (int j = 0; j < 4; ++j) {
    float v = bf2f(in[base + (j >> 1) * Win + (j & 1)]);
    v = fmaxf((v - m) * is * gg + bb, 0.f);
    best = fmaxf(best, v);
  }
  out[idx] = f2bf(best);
}

__global__ __launch_bounds__(256) void bn_relu(const unsigned short* __restrict__ in,
                                               const float* __restrict__ stats,
                                               const float* __restrict__ g,
                                               const float* __restrict__ bta,
                                               unsigned short* __restrict__ out,
                                               int C, int HW, int total) {
  int idx = blockIdx.x * 256 + threadIdx.x;
  if (idx >= total) return;
  int c = (idx / HW) % C;
  float v = bf2f(in[idx]);
  v = fmaxf((v - stats[2 * c]) * stats[2 * c + 1] * g[c] + bta[c], 0.f);
  out[idx] = f2bf(v);
}

// =====================================================================
// heads: one WMMA conv, Cin=256, Cout padded to 32 (21|3|2 routed to
// d_out sections, f32 + bias). block tile: 64 pixels x 32 channels.
// =====================================================================
__global__ __launch_bounds__(256) void conv3x3_heads_wmma(const unsigned short* __restrict__ in,
                                                          const unsigned short* __restrict__ wt,
                                                          const float* __restrict__ hbias,
                                                          float* __restrict__ out,
                                                          int H, int W) {
  const int Cin = 256;
  __shared__ __align__(16) unsigned short sIn[3 * 68 * 40];  // [row][pix<66][ci]
  __shared__ __align__(16) unsigned short sB[9 * 32 * 40];   // [tap][co][ci]

  const int tid   = threadIdx.x;
  const int lane  = tid & 31;
  const int wave  = tid >> 5;
  const int waveN = wave & 1;
  const int waveM = wave >> 1;
  const int l16   = lane & 15;
  const int lhalf = lane >> 4;

  const int xbase = blockIdx.x * 64;
  const int y     = blockIdx.y;
  const int b     = blockIdx.z;

  v8f acc = {0.f, 0.f, 0.f, 0.f, 0.f, 0.f, 0.f, 0.f};

  #pragma unroll 1
  for (int cic = 0; cic < (Cin >> 5); ++cic) {
    const int c0 = cic << 5;

    for (int idx = tid; idx < 3 * 66 * 32; idx += 256) {
      const int row = idx / (66 * 32);
      const int rem = idx % (66 * 32);
      const int ci  = rem / 66;
      const int pix = rem % 66;
      const int iy  = y + row - 1;
      const int gx  = xbase + pix - 1;
      unsigned short v = 0;
      if ((unsigned)iy < (unsigned)H && (unsigned)gx < (unsigned)W)
        v = in[(((size_t)b * Cin + (c0 + ci)) * H + iy) * W + gx];
      sIn[(row * 68 + pix) * 40 + ci] = v;
    }

    {
      const unsigned short* wsrc = wt + (size_t)cic * (9 * 32 * 32);
      for (int g = tid; g < 9 * 32 * 4; g += 256) {       // 1152 groups
        const int tap = g >> 7;                           // g / 128
        const int r   = g & 127;
        const int co  = r >> 2;
        const int cib = (r & 3) << 3;
        async_copy_b128(lds_off32(&sB[(tap * 32 + co) * 40 + cib]),
                        wsrc + ((size_t)(tap * 32 + co) * 32 + cib));
      }
      wait_async0();
    }
    __syncthreads();

    #pragma unroll
    for (int tap = 0; tap < 9; ++tap) {
      const int dy = tap / 3;
      const int dx = tap % 3;
      FragBF a, bm;
      const int pixIdx = waveM * 16 + l16 + dx;           // 0..65
      const int abase  = (dy * 68 + pixIdx) * 40 + lhalf * 8;
      a.h[0] = *(const u16x8*)&sIn[abase];
      a.h[1] = *(const u16x8*)&sIn[abase + 16];
      const int cor = waveN * 16 + l16;
      const int kb  = lhalf * 8;
      bm.h[0] = *(const u16x8*)&sB[(tap * 32 + cor) * 40 + kb];
      bm.h[1] = *(const u16x8*)&sB[(tap * 32 + cor) * 40 + kb + 16];
      acc = __builtin_amdgcn_wmma_f32_16x16x32_bf16(false, a.v, false, bm.v,
                                                    (short)0, acc, false, false);
    }
    __syncthreads();
  }

  const int HW = H * W;
  const int O1 = 4 * 21 * HW;
  const int O2 = O1 + 4 * 3 * HW;
  const int co = waveN * 16 + l16;
  if (co < 26) {
    const float bias = hbias[co];
    #pragma unroll
    for (int r = 0; r < 8; ++r) {
      const int x = xbase + waveM * 16 + r + lhalf * 8;
      if (x < W) {
        const size_t pix = (size_t)y * W + x;
        const float v = acc[r] + bias;
        if      (co < 21) out[((size_t)(b * 21 + co)) * HW + pix] = v;
        else if (co < 24) out[O1 + ((size_t)(b * 3 + (co - 21))) * HW + pix] = v;
        else              out[O2 + ((size_t)(b * 2 + (co - 24))) * HW + pix] = v;
      }
    }
  }
}

// =====================================================================
// launcher
// =====================================================================
extern "C" void kernel_launch(void* const* d_in, const int* in_sizes, int n_in,
                              void* d_out, int out_size, void* d_ws, size_t ws_size,
                              hipStream_t stream) {
  const float* pillars = (const float*)d_in[0];
  const int*   coords  = (const int*)  d_in[1];
  const float* lin_w   = (const float*)d_in[2];
  const float* bn0_g   = (const float*)d_in[4];
  const float* bn0_b   = (const float*)d_in[5];
  const float* c1_w    = (const float*)d_in[6];
  const float* bn1_g   = (const float*)d_in[8];
  const float* bn1_b   = (const float*)d_in[9];
  const float* c2_w    = (const float*)d_in[10];
  const float* bn2_g   = (const float*)d_in[12];
  const float* bn2_b   = (const float*)d_in[13];
  const float* h1_w    = (const float*)d_in[14];
  const float* hbn_g   = (const float*)d_in[16];
  const float* hbn_b   = (const float*)d_in[17];
  const float* box_w   = (const float*)d_in[18];
  const float* box_b   = (const float*)d_in[19];
  const float* cls_w   = (const float*)d_in[20];
  const float* cls_b   = (const float*)d_in[21];
  const float* dir_w   = (const float*)d_in[22];
  const float* dir_b   = (const float*)d_in[23];

  const long ROWS = (long)BN_ * P_ * NPT_;              // 1,536,000
  const int  HW0 = H0_ * W0_, HW1 = H1_ * W1_, HW2 = H2_ * W2_;

  auto al = [](size_t x) { return (x + 255) & ~(size_t)255; };
  const size_t szA = (size_t)BN_ * 64 * HW0 * 2;        // 227 MB
  const size_t szB = szA;                               // 227 MB
  size_t off = 0;
  char* base = (char*)d_ws;
  unsigned short* bufA  = (unsigned short*)(base + off); off = al(off + szA);
  unsigned short* bufB  = (unsigned short*)(base + off); off = al(off + szB);
  unsigned short* feats = (unsigned short*)(base + off); off = al(off + (size_t)BN_ * P_ * 64 * 2);
  unsigned short* w1b   = (unsigned short*)(base + off); off = al(off + (size_t)9 * 64 * 64 * 2);
  unsigned short* w2b   = (unsigned short*)(base + off); off = al(off + (size_t)9 * 128 * 64 * 2);
  unsigned short* whb   = (unsigned short*)(base + off); off = al(off + (size_t)9 * 256 * 128 * 2);
  unsigned short* whd   = (unsigned short*)(base + off); off = al(off + (size_t)9 * 32 * 256 * 2);
  float* hbias = (float*)(base + off); off = al(off + 32 * 4);
  float* stats = (float*)(base + off); off = al(off + 512 * 4);
  float* part  = (float*)(base + off); off = al(off + (size_t)256 * SEG_ * 2 * 4);

  // ---- weights -> blocked bf16 ----
  convert_w<<<(64 * 64 * 9 + 255) / 256, 256, 0, stream>>>(c1_w, w1b, 64, 64);
  convert_w<<<(128 * 64 * 9 + 255) / 256, 256, 0, stream>>>(c2_w, w2b, 128, 64);
  convert_w<<<(256 * 128 * 9 + 255) / 256, 256, 0, stream>>>(h1_w, whb, 256, 128);
  convert_heads<<<(9 * 32 * 256 + 255) / 256, 256, 0, stream>>>(box_w, cls_w, dir_w,
                                                                box_b, cls_b, dir_b, whd, hbias);

  // ---- stage 0: linear -> bn0 stats -> bn+relu+max -> feats ----
  linear_kernel<<<(int)((ROWS + 255) / 256), 256, 0, stream>>>(pillars, lin_w, bufB, ROWS);
  stats_partial_rows<<<dim3(64, SEG_), 256, 0, stream>>>(bufB, part, ROWS);
  stats_finalize<<<64, 64, 0, stream>>>(part, stats, (float)ROWS);
  bnmax_kernel<<<(BN_ * P_ * 64 + 255) / 256, 256, 0, stream>>>(bufB, stats, bn0_g, bn0_b,
                                                                feats, BN_ * P_ * 64);

  // ---- scatter into pseudo image (bufA) ----
  zero_u32<<<4096, 256, 0, stream>>>((unsigned int*)bufA, szA / 4);
  scatter_kernel<<<(BN_ * P_ * 64 + 255) / 256, 256, 0, stream>>>(feats, coords, bufA,
                                                                  BN_ * P_ * 64);

  // ---- conv1 (64->64 @ 666x666) + bn1 + relu + pool -> 333x333 ----
  conv3x3_wmma<64><<<dim3((W0_ + 31) / 32, H0_, BN_ * 1), 256, 0, stream>>>(bufA, w1b, bufB,
                                                                            64, H0_, W0_);
  stats_partial_nchw<<<dim3(64, SEG_), 256, 0, stream>>>(bufB, part, 64, HW0, BN_);
  stats_finalize<<<64, 64, 0, stream>>>(part, stats, (float)((long)BN_ * HW0));
  {
    int total = BN_ * 64 * HW1;
    bn_relu_pool<<<(total + 255) / 256, 256, 0, stream>>>(bufB, stats, bn1_g, bn1_b, bufA,
                                                          64, H0_, W0_, H1_, W1_, total);
  }

  // ---- conv2 (64->128 @ 333x333) + bn2 + relu + pool -> 166x166 ----
  conv3x3_wmma<64><<<dim3((W1_ + 31) / 32, H1_, BN_ * 2), 256, 0, stream>>>(bufA, w2b, bufB,
                                                                            128, H1_, W1_);
  stats_partial_nchw<<<dim3(128, SEG_), 256, 0, stream>>>(bufB, part, 128, HW1, BN_);
  stats_finalize<<<128, 64, 0, stream>>>(part, stats, (float)((long)BN_ * HW1));
  {
    int total = BN_ * 128 * HW2;
    bn_relu_pool<<<(total + 255) / 256, 256, 0, stream>>>(bufB, stats, bn2_g, bn2_b, bufA,
                                                          128, H1_, W1_, H2_, W2_, total);
  }

  // ---- h1 (128->256 @ 166x166) + hbn + relu ----
  conv3x3_wmma<128><<<dim3((W2_ + 31) / 32, H2_, BN_ * 4), 256, 0, stream>>>(bufA, whb, bufB,
                                                                             256, H2_, W2_);
  stats_partial_nchw<<<dim3(256, SEG_), 256, 0, stream>>>(bufB, part, 256, HW2, BN_);
  stats_finalize<<<256, 64, 0, stream>>>(part, stats, (float)((long)BN_ * HW2));
  {
    int total = BN_ * 256 * HW2;
    bn_relu<<<(total + 255) / 256, 256, 0, stream>>>(bufB, stats, hbn_g, hbn_b, bufA,
                                                     256, HW2, total);
  }

  // ---- heads (256 -> 21|3|2, fused, padded to 32) -> d_out (f32) ----
  conv3x3_heads_wmma<<<dim3((W2_ + 63) / 64, H2_, BN_), 256, 0, stream>>>(bufA, whd, hbias,
                                                                          (float*)d_out,
                                                                          H2_, W2_);
}
